// TopoformerLayer_56650618635160
// MI455X (gfx1250) — matrix operand order repaired and checked by reading, not compiled, not run on hardware
//
#include <hip/hip_runtime.h>
#include <math.h>

// ---------------------------------------------------------------------------
// CDNA5 (gfx1250) WMMA types & fragment helpers (wave32)
// ---------------------------------------------------------------------------
typedef __attribute__((ext_vector_type(16))) _Float16 v16h;
typedef __attribute__((ext_vector_type(8)))  _Float16 v8h;
typedef __attribute__((ext_vector_type(8)))  float    v8f;
typedef __attribute__((ext_vector_type(4)))  unsigned int v4u;
typedef __attribute__((ext_vector_type(8)))  int      v8i;
typedef __attribute__((ext_vector_type(4)))  int      v4i;

#if __has_builtin(__builtin_amdgcn_tensor_load_to_lds)
#define USE_TDM 1
#else
#define USE_TDM 0
#endif

union Frag16 {            // 16 halves = one WMMA A or B operand (wave32)
    v16h v;
    v8h  h[2];
    _Float16 e[16];
};

static __device__ __forceinline__ v8f wmma_f16(const Frag16& a, const Frag16& b, v8f c) {
    return __builtin_amdgcn_wmma_f32_16x16x32_f16(false, a.v, false, b.v,
                                                  (short)0, c, false, false);
}

static __device__ __forceinline__ v8h pack8(float4 a, float4 b, float sc) {
    v8h r;
    r[0] = (_Float16)(a.x * sc); r[1] = (_Float16)(a.y * sc);
    r[2] = (_Float16)(a.z * sc); r[3] = (_Float16)(a.w * sc);
    r[4] = (_Float16)(b.x * sc); r[5] = (_Float16)(b.y * sc);
    r[6] = (_Float16)(b.z * sc); r[7] = (_Float16)(b.w * sc);
    return r;
}

// ---------------------------------------------------------------------------
// Generic tiled WMMA GEMM: C[M,N] = epi(A[M,K] * W[K,N] + bias)
//   block tile 128x64, 256 threads = 8 waves, wave tile 32x32 (4 WMMA/k-step)
//   A tile DMA'd into LDS by the Tensor Data Mover (f32), B tile converted to
//   f16 in LDS by the waves; A-fragments convert f32->f16 at read time.
// epilogue: 0 = bias, 1 = bias + exact-erf GELU, 2 = dist (sq[m]+sq[n]-2*acc)
// accumulate: C += result (bias may be null)
// ---------------------------------------------------------------------------
#define GBM 128
#define GBN 64
#define GBK 32

__global__ __launch_bounds__(256) void gemm_f16_wmma(
    const float* __restrict__ A, int lda, long strideA,
    const float* __restrict__ W, int ldb, long strideB, int transB,
    const float* __restrict__ bias,
    float* __restrict__ C, int ldc, long strideC,
    int M, int N, int Kd,
    int epilogue, int accumulate,
    const float* __restrict__ sqp, long sqStride)
{
#if USE_TDM
    __shared__ __align__(16) float    As32[GBM * GBK];  // [m][k] f32 (TDM dest)
#else
    __shared__ __align__(16) _Float16 As16[GBM * GBK];  // [m][k] f16
#endif
    __shared__ __align__(16) _Float16 Bs[GBN * GBK];    // [n][k]  (k-major per row)

    const int tid  = threadIdx.x;
    const int lane = tid & 31;
    const int wave = tid >> 5;
    const int ln15 = lane & 15;
    const int hi   = lane >> 4;
    const int wm   = wave >> 1;    // 0..3 -> M sub-block
    const int wn   = wave & 1;     // 0..1 -> N sub-block
    const int z    = blockIdx.z;
    const int m0   = blockIdx.x * GBM;
    const int n0   = blockIdx.y * GBN;

    A += (long)z * strideA;
    W += (long)z * strideB;
    C += (long)z * strideC;

    v8f acc[2][2];
    v8f zero = {};
#pragma unroll
    for (int s = 0; s < 2; ++s)
#pragma unroll
        for (int t = 0; t < 2; ++t) acc[s][t] = zero;

    for (int k0 = 0; k0 < Kd; k0 += GBK) {
#if USE_TDM
        // ---- A tile via Tensor Data Mover: 128 rows x 32 f32, stride lda ----
        if (wave == 0) {
            unsigned long long ga =
                (unsigned long long)(const void*)(A + (long)m0 * lda + k0);
            unsigned ldsa = (unsigned)(unsigned long long)(const void*)As32;
            v4u g0;
            g0[0] = 1u;                                   // count = 1 (valid D#)
            g0[1] = ldsa;                                 // lds_addr
            g0[2] = (unsigned)(ga & 0xFFFFFFFFu);         // global_addr lo
            g0[3] = (unsigned)((ga >> 32) & 0x1FFFFFFu)   // global_addr hi
                    | (2u << 30);                         // type = 2 ("image")
            v8i g1;
            g1[0] = (int)(2u << 16);      // data_size = 4B
            g1[1] = (int)(32u << 16);     // tensor_dim0 = 32 (lo16 @ bits 63:48)
            g1[2] = (int)(128u << 16);    // tensor_dim0 hi=0 | tensor_dim1 = 128
            g1[3] = (int)(32u << 16);     // tensor_dim1 hi=0 | tile_dim0 = 32
            g1[4] = (int)128u;            // tile_dim1 = 128, tile_dim2 = 0
            g1[5] = (int)(unsigned)lda;   // tensor_dim0_stride lo32 (elements)
            g1[6] = 0;                    // stride0 hi | stride1 lo
            g1[7] = 0;                    // stride1 hi
            v4i gz = {0, 0, 0, 0};
#if defined(__clang_major__) && (__clang_major__ >= 23)
            v8i gz8 = {0, 0, 0, 0, 0, 0, 0, 0};
            __builtin_amdgcn_tensor_load_to_lds(g0, g1, gz, gz, gz8, 0);
#else
            __builtin_amdgcn_tensor_load_to_lds(g0, g1, gz, gz, 0);
#endif
        }
#else
        // ---- A tile: 128x32 f32 -> f16, 16 elems/thread ----
#pragma unroll
        for (int j = 0; j < 4; ++j) {
            int idx = tid * 16 + j * 4;
            int r = idx >> 5, c = idx & 31;
            float4 f = *(const float4*)(A + (long)(m0 + r) * lda + k0 + c);
            _Float16* d = &As16[r * GBK + c];
            d[0] = (_Float16)f.x; d[1] = (_Float16)f.y;
            d[2] = (_Float16)f.z; d[3] = (_Float16)f.w;
        }
#endif
        // ---- B tile: store as [n][k] f16 ----
        if (!transB) {
#pragma unroll
            for (int j = 0; j < 2; ++j) {
                int idx = tid * 8 + j * 4;
                int kk = idx >> 6, n = idx & 63;
                float4 f = *(const float4*)(W + (long)(k0 + kk) * ldb + n0 + n);
                Bs[(n + 0) * GBK + kk] = (_Float16)f.x;
                Bs[(n + 1) * GBK + kk] = (_Float16)f.y;
                Bs[(n + 2) * GBK + kk] = (_Float16)f.z;
                Bs[(n + 3) * GBK + kk] = (_Float16)f.w;
            }
        } else {
#pragma unroll
            for (int j = 0; j < 2; ++j) {
                int idx = tid * 8 + j * 4;
                int n = idx >> 5, kk = idx & 31;
                float4 f = *(const float4*)(W + (long)(n0 + n) * ldb + k0 + kk);
                _Float16* d = &Bs[n * GBK + kk];
                d[0] = (_Float16)f.x; d[1] = (_Float16)f.y;
                d[2] = (_Float16)f.z; d[3] = (_Float16)f.w;
            }
        }
        if (k0 + GBK < Kd)   // pull next B rows toward L2/WGP$
            __builtin_prefetch(W + (transB ? ((long)(n0 + (tid & 63)) * ldb + k0 + GBK)
                                           : ((long)(k0 + GBK + (tid & 31)) * ldb + n0)),
                               0, 1);
#if USE_TDM
        if (wave == 0) {
#if __has_builtin(__builtin_amdgcn_s_wait_tensorcnt)
            __builtin_amdgcn_s_wait_tensorcnt(0);
#else
            asm volatile("s_wait_tensorcnt 0x0" ::: "memory");
#endif
        }
#endif
        __syncthreads();

        // ---- fragments (ISA 16-bit A 16x32 & k-major B layouts) ----
        Frag16 af[2], bf[2];
#pragma unroll
        for (int s = 0; s < 2; ++s) {
            int r = wm * 32 + s * 16 + ln15;
#if USE_TDM
            const float* ap = &As32[r * GBK];
            af[s].h[0] = pack8(*(const float4*)(ap + hi * 8),
                               *(const float4*)(ap + hi * 8 + 4), 1.0f);
            af[s].h[1] = pack8(*(const float4*)(ap + 16 + hi * 8),
                               *(const float4*)(ap + 16 + hi * 8 + 4), 1.0f);
#else
            af[s].h[0] = *(const v8h*)&As16[r * GBK + hi * 8];
            af[s].h[1] = *(const v8h*)&As16[r * GBK + 16 + hi * 8];
#endif
        }
#pragma unroll
        for (int t = 0; t < 2; ++t) {
            int n = wn * 32 + t * 16 + ln15;
            bf[t].h[0] = *(const v8h*)&Bs[n * GBK + hi * 8];       // K = hi*8 + 0..7
            bf[t].h[1] = *(const v8h*)&Bs[n * GBK + 16 + hi * 8];  // K = 16 + hi*8 + 0..7
        }
#pragma unroll
        for (int s = 0; s < 2; ++s)
#pragma unroll
            for (int t = 0; t < 2; ++t)
                acc[s][t] = wmma_f16(af[s], bf[t], acc[s][t]);
        __syncthreads();
    }

    // ---- epilogue: C layout row = v + hi*8, col = lane&15 ----
#pragma unroll
    for (int s = 0; s < 2; ++s) {
#pragma unroll
        for (int t = 0; t < 2; ++t) {
            int row0 = m0 + wm * 32 + s * 16 + hi * 8;
            int col  = n0 + wn * 32 + t * 16 + ln15;
            float bcol = bias ? bias[col] : 0.0f;
            float sqc  = (epilogue == 2) ? sqp[z * sqStride + col] : 0.0f;
#pragma unroll
            for (int v = 0; v < 8; ++v) {
                float x = acc[s][t][v];
                long off = (long)(row0 + v) * ldc + col;
                float o;
                if (epilogue == 2) {
                    o = sqp[z * sqStride + row0 + v] + sqc - 2.0f * x;
                } else {
                    o = x + bcol;
                    if (accumulate) o += C[off];
                    if (epilogue == 1)
                        o = 0.5f * o * (1.0f + erff(o * 0.70710678118654752f));
                }
                C[off] = o;
            }
        }
    }
}

// ---------------------------------------------------------------------------
// sq[row] = sum_k tp[row,k]^2
// ---------------------------------------------------------------------------
__global__ __launch_bounds__(128) void rowsumsq_kernel(const float* __restrict__ tp,
                                                       float* __restrict__ sq, int Kd)
{
    __shared__ float red[128];
    int row = blockIdx.x;
    const float* p = tp + (long)row * Kd;
    float s = 0.0f;
    for (int i = threadIdx.x; i < Kd; i += 128) { float v = p[i]; s += v * v; }
    red[threadIdx.x] = s; __syncthreads();
    for (int o = 64; o > 0; o >>= 1) {
        if (threadIdx.x < o) red[threadIdx.x] += red[threadIdx.x + o];
        __syncthreads();
    }
    if (threadIdx.x == 0) sq[row] = red[0];
}

// ---------------------------------------------------------------------------
// top-K (=32) smallest per row of d2; one wave per row, 8 rows/block.
// jax top_k tie-break: first occurrence -> keep smallest index on ties.
// ---------------------------------------------------------------------------
__global__ __launch_bounds__(256) void topk_kernel(const float* __restrict__ d2,
                                                   int* __restrict__ idxo, int S, int K)
{
    __shared__ float rows[8][1024];
    const int wave = threadIdx.x >> 5;
    const int lane = threadIdx.x & 31;
    const int row  = blockIdx.x * 8 + wave;          // b*S + i
    const int b    = row / S;
    const int i    = row - b * S;
    const float* src = d2 + (long)b * S * S + (long)i * S;
    float* rw = rows[wave];
    for (int t = 0; t < 32; ++t) rw[lane + 32 * t] = src[lane + 32 * t];

    float mv = __builtin_inff(); int mj = 0x7fffffff;
    for (int t = 0; t < 32; ++t) {
        int j = lane + 32 * t; float v = rw[j];
        if (v < mv || (v == mv && j < mj)) { mv = v; mj = j; }
    }
    for (int t = 0; t < K; ++t) {
        float bv = mv; int bj = mj;
        for (int m = 16; m >= 1; m >>= 1) {
            float ov = __shfl_xor(bv, m, 32);
            int   oj = __shfl_xor(bj, m, 32);
            if (ov < bv || (ov == bv && oj < bj)) { bv = ov; bj = oj; }
        }
        if (lane == 0) idxo[(long)row * K + t] = bj;
        if ((bj & 31) == lane) {          // this lane owns the removed element
            rw[bj] = __builtin_inff();
            mv = __builtin_inff(); mj = 0x7fffffff;
            for (int q = 0; q < 32; ++q) {
                int j = lane + 32 * q; float v = rw[j];
                if (v < mv || (v == mv && j < mj)) { mv = v; mj = j; }
            }
        }
    }
}

// ---------------------------------------------------------------------------
// pooled[row,:] = mean_k x[b, idx[row,k], :]
// ---------------------------------------------------------------------------
__global__ __launch_bounds__(256) void pool_kernel(const float* __restrict__ x,
                                                   const int* __restrict__ idx,
                                                   float* __restrict__ pooled,
                                                   int S, int D, int K)
{
    __shared__ int nb[32];
    const int row = blockIdx.x;            // b*S + i
    const int b   = row / S;
    if (threadIdx.x < K) nb[threadIdx.x] = idx[(long)row * K + threadIdx.x];
    __syncthreads();
    const float* xb = x + (long)b * S * D;
    const float inv = 1.0f / (float)K;
    for (int c = threadIdx.x; c < (D >> 2); c += 256) {
        float sx = 0, sy = 0, sz = 0, sw = 0;
        for (int k = 0; k < K; ++k) {
            float4 v = *(const float4*)(xb + (long)nb[k] * D + c * 4);
            sx += v.x; sy += v.y; sz += v.z; sw += v.w;
        }
        float4 o; o.x = sx * inv; o.y = sy * inv; o.z = sz * inv; o.w = sw * inv;
        *(float4*)(pooled + (long)row * D + c * 4) = o;
    }
}

// ---------------------------------------------------------------------------
// LayerNorm over D=768 (3 elems/thread @ 256 threads), optional residual add
// and optional ReLU.  Safe in-place.
// ---------------------------------------------------------------------------
__global__ __launch_bounds__(256) void ln_kernel(const float* __restrict__ A,
                                                 const float* __restrict__ Badd,
                                                 const float* __restrict__ gamma,
                                                 const float* __restrict__ beta,
                                                 float* __restrict__ outp,
                                                 int D, int doRelu)
{
    __shared__ float red[256];
    const int row = blockIdx.x, tid = threadIdx.x;
    const float* a  = A + (long)row * D;
    const float* bb = Badd ? Badd + (long)row * D : nullptr;
    float vals[3];
    float s = 0.0f;
#pragma unroll
    for (int j = 0; j < 3; ++j) {
        int c = tid + j * 256;
        float v = a[c];
        if (bb) v += bb[c];
        vals[j] = v; s += v;
    }
    red[tid] = s; __syncthreads();
    for (int o = 128; o > 0; o >>= 1) { if (tid < o) red[tid] += red[tid + o]; __syncthreads(); }
    float mu = red[0] / (float)D;
    __syncthreads();
    float s2 = 0.0f;
#pragma unroll
    for (int j = 0; j < 3; ++j) { float d = vals[j] - mu; s2 += d * d; }
    red[tid] = s2; __syncthreads();
    for (int o = 128; o > 0; o >>= 1) { if (tid < o) red[tid] += red[tid + o]; __syncthreads(); }
    float rstd = rsqrtf(red[0] / (float)D + 1e-5f);
#pragma unroll
    for (int j = 0; j < 3; ++j) {
        int c = tid + j * 256;
        float o = (vals[j] - mu) * rstd * gamma[c] + beta[c];
        if (doRelu) o = fmaxf(o, 0.0f);
        outp[(long)row * D + c] = o;
    }
}

// ---------------------------------------------------------------------------
// comb = ao*sigmoid(gp) + topo*(1-sigmoid(gp))
// ---------------------------------------------------------------------------
__global__ __launch_bounds__(256) void gate_combine_kernel(const float* __restrict__ gp,
                                                           const float* __restrict__ ao,
                                                           const float* __restrict__ topo,
                                                           float* __restrict__ comb, long n)
{
    long i = (long)blockIdx.x * 256 + threadIdx.x;
    if (i < n) {
        float g = 1.0f / (1.0f + __expf(-gp[i]));
        comb[i] = ao[i] * g + topo[i] * (1.0f - g);
    }
}

// ---------------------------------------------------------------------------
// Flash attention: one (b,h) per blockIdx.{z,y}, 128 q-rows/block, 16 q-rows
// per wave.  K/V tiles are staged ONCE per block into LDS (f16, V transposed)
// since all 8 waves consume the same tiles; QK^T and PV via WMMA; P is
// re-laid-out through a per-wave LDS tile with an s_wait_dscnt fence.
// ---------------------------------------------------------------------------
__global__ __launch_bounds__(256) void attn_kernel(const float* __restrict__ Q,
                                                   const float* __restrict__ Kf,
                                                   const float* __restrict__ V,
                                                   float* __restrict__ O,
                                                   int S, int D, int HD)
{
    __shared__ __align__(16) _Float16 Ks[32 * 64];      // [j][d]  (4 KB)
    __shared__ __align__(16) _Float16 Vs[64 * 32];      // [d][j]  (4 KB)
    __shared__ __align__(16) _Float16 Pl[8][16 * 32];   // per-wave P tile (8 KB)
    const int wave = threadIdx.x >> 5;
    const int lane = threadIdx.x & 31;
    const int ln15 = lane & 15;
    const int hi   = lane >> 4;
    const int h    = blockIdx.y;
    const int b    = blockIdx.z;
    const int i0   = blockIdx.x * 128 + wave * 16;
    const long base = (long)b * S * D + h * HD;
    const float qscale = rsqrtf((float)HD);

    // Q fragments (A-layout), scaled by 1/sqrt(HD)
    Frag16 qa[2];
    {
        const float* qrow = Q + base + (long)(i0 + ln15) * D;
#pragma unroll
        for (int ks = 0; ks < 2; ++ks)
#pragma unroll
            for (int c = 0; c < 2; ++c) {
                int kb = ks * 32 + c * 16 + hi * 8;
                float4 f0 = *(const float4*)(qrow + kb);
                float4 f1 = *(const float4*)(qrow + kb + 4);
                qa[ks].h[c] = pack8(f0, f1, qscale);
            }
    }

    float mrow[8], lrow[8];
    v8f o[4];
    v8f zero = {};
#pragma unroll
    for (int v = 0; v < 8; ++v) { mrow[v] = -__builtin_inff(); lrow[v] = 0.0f; }
#pragma unroll
    for (int d = 0; d < 4; ++d) o[d] = zero;

    for (int j0 = 0; j0 < S; j0 += 32) {
        // ---- cooperative K/V tile staging (2048 halves each, 8/thread) ----
        {
            int idx = threadIdx.x * 8;
            int jj = idx >> 6, d0 = idx & 63;
            const float* kp = Kf + base + (long)(j0 + jj) * D + d0;
            float4 ka = *(const float4*)kp;
            float4 kb = *(const float4*)(kp + 4);
            *(v8h*)&Ks[jj * 64 + d0] = pack8(ka, kb, 1.0f);
            const float* vp = V + base + (long)(j0 + jj) * D + d0;
            float4 va = *(const float4*)vp;
            float4 vb = *(const float4*)(vp + 4);
            v8h vh = pack8(va, vb, 1.0f);
#pragma unroll
            for (int i = 0; i < 8; ++i) Vs[(d0 + i) * 32 + jj] = vh[i];
        }
        __syncthreads();

        // ---- S = q k^T (two 16x16 tiles over j) ----
        Frag16 bk[2][2];
#pragma unroll
        for (int t = 0; t < 2; ++t) {
            const _Float16* kr = &Ks[(t * 16 + ln15) * 64];
#pragma unroll
            for (int ks = 0; ks < 2; ++ks) {
                bk[t][ks].h[0] = *(const v8h*)(kr + ks * 32 + hi * 8);
                bk[t][ks].h[1] = *(const v8h*)(kr + ks * 32 + 16 + hi * 8);
            }
        }
        v8f sc[2];
#pragma unroll
        for (int t = 0; t < 2; ++t) {
            sc[t] = zero;
            sc[t] = wmma_f16(qa[0], bk[t][0], sc[t]);
            sc[t] = wmma_f16(qa[1], bk[t][1], sc[t]);
        }
        // ---- online softmax (per row v + hi*8, 16-lane groups) ----
#pragma unroll
        for (int v = 0; v < 8; ++v) {
            float a0 = sc[0][v], a1 = sc[1][v];
            float bm = fmaxf(a0, a1);
            bm = fmaxf(bm, __shfl_xor(bm, 1, 32));
            bm = fmaxf(bm, __shfl_xor(bm, 2, 32));
            bm = fmaxf(bm, __shfl_xor(bm, 4, 32));
            bm = fmaxf(bm, __shfl_xor(bm, 8, 32));
            float mn = fmaxf(mrow[v], bm);
            float scale = __expf(mrow[v] - mn);
            float p0 = __expf(a0 - mn), p1 = __expf(a1 - mn);
            float rs = p0 + p1;
            rs += __shfl_xor(rs, 1, 32);
            rs += __shfl_xor(rs, 2, 32);
            rs += __shfl_xor(rs, 4, 32);
            rs += __shfl_xor(rs, 8, 32);
            lrow[v] = lrow[v] * scale + rs;
            mrow[v] = mn;
            sc[0][v] = p0; sc[1][v] = p1;
#pragma unroll
            for (int d = 0; d < 4; ++d) o[d][v] *= scale;
        }
        // ---- P -> LDS (C-layout) -> A-fragment relayout (per-wave) ----
        _Float16* pw = Pl[wave];
#pragma unroll
        for (int t = 0; t < 2; ++t)
#pragma unroll
            for (int v = 0; v < 8; ++v)
                pw[(v + hi * 8) * 32 + t * 16 + ln15] = (_Float16)sc[t][v];
        asm volatile("s_wait_dscnt 0" ::: "memory");   // intra-wave DS fence
        Frag16 pa;
        pa.h[0] = *(const v8h*)&pw[ln15 * 32 + hi * 8];
        pa.h[1] = *(const v8h*)&pw[ln15 * 32 + 16 + hi * 8];
        // ---- O += P V  (B-fragments from transposed V tile in LDS) ----
#pragma unroll
        for (int dsub = 0; dsub < 4; ++dsub) {
            Frag16 bv;
            const _Float16* vr = &Vs[(dsub * 16 + ln15) * 32];
            bv.h[0] = *(const v8h*)(vr + hi * 8);
            bv.h[1] = *(const v8h*)(vr + 16 + hi * 8);
            o[dsub] = wmma_f16(pa, bv, o[dsub]);
        }
        __syncthreads();   // Ks/Vs reused next iteration
    }

    // ---- normalize & write ----
    float* orow = O + base;
#pragma unroll
    for (int dsub = 0; dsub < 4; ++dsub)
#pragma unroll
        for (int v = 0; v < 8; ++v) {
            int i = v + hi * 8;
            orow[(long)(i0 + i) * D + dsub * 16 + ln15] = o[dsub][v] / lrow[v];
        }
}

// ---------------------------------------------------------------------------
// Host orchestration
// ---------------------------------------------------------------------------
extern "C" void kernel_launch(void* const* d_in, const int* in_sizes, int n_in,
                              void* d_out, int out_size, void* d_ws, size_t ws_size,
                              hipStream_t stream)
{
    (void)in_sizes; (void)n_in; (void)out_size; (void)ws_size;
    const float* x   = (const float*)d_in[0];
    const float* Wp  = (const float*)d_in[1];
    const float* bp  = (const float*)d_in[2];
    const float* We  = (const float*)d_in[3];
    const float* be  = (const float*)d_in[4];
    const float* ge  = (const float*)d_in[5];
    const float* bbe = (const float*)d_in[6];
    const float* Wq  = (const float*)d_in[7];
    const float* bq  = (const float*)d_in[8];
    const float* Wk  = (const float*)d_in[9];
    const float* bk  = (const float*)d_in[10];
    const float* Wv  = (const float*)d_in[11];
    const float* bv  = (const float*)d_in[12];
    const float* Wo  = (const float*)d_in[13];
    const float* bo  = (const float*)d_in[14];
    const float* Wg  = (const float*)d_in[15];
    const float* bg  = (const float*)d_in[16];
    const float* g1  = (const float*)d_in[17];
    const float* b1  = (const float*)d_in[18];
    const float* g2  = (const float*)d_in[19];
    const float* b2  = (const float*)d_in[20];
    const float* Wf1 = (const float*)d_in[21];
    const float* bf1 = (const float*)d_in[22];
    const float* Wf2 = (const float*)d_in[23];
    const float* bf2 = (const float*)d_in[24];
    float* out = (float*)d_out;

    const int Bb = 4, S = 1024, Dd = 768, H = 12, HD = 64, Kk = 32;
    const int NS = Bb * S;           // 4096
    const int Dh = Dd / 2;           // 384
    const int Df = 4 * Dd;           // 3072

    float* ws = (float*)d_ws;
    size_t off = 0;
    auto alloc = [&](size_t n) { float* p = ws + off; off += (n + 3) & ~(size_t)3; return p; };
    float* tp     = alloc((size_t)NS * Dh);
    float* sq     = alloc((size_t)NS);
    float* d2     = alloc((size_t)Bb * S * S);
    int*   idxb   = (int*)alloc((size_t)NS * Kk);
    float* pooled = alloc((size_t)NS * Dd);
    float* topo   = alloc((size_t)NS * Dd);
    float* q      = alloc((size_t)NS * Dd);
    float* kbuf   = alloc((size_t)NS * Dd);
    float* vbuf   = alloc((size_t)NS * Dd);
    float* ao     = alloc((size_t)NS * Dd);
    float* gp     = alloc((size_t)NS * Dd);
    float* comb   = alloc((size_t)NS * Dd);
    float* atn    = alloc((size_t)NS * Dd);
    float* hbuf   = alloc((size_t)NS * Dd);
    float* f1     = alloc((size_t)NS * Df);
    float* f2     = alloc((size_t)NS * Dd);

    dim3 blk(256);
    auto gemm = [&](const float* A, int lda, long sA,
                    const float* W, int ldb, long sB, int tB,
                    const float* bias, float* C, int ldc, long sC,
                    int M, int N, int Kd, int epi, int acc,
                    const float* sqpv, long sqs, int batch) {
        dim3 grid(M / GBM, N / GBN, batch);
        gemm_f16_wmma<<<grid, blk, 0, stream>>>(A, lda, sA, W, ldb, sB, tB, bias,
                                                C, ldc, sC, M, N, Kd, epi, acc, sqpv, sqs);
    };

    // ---- topology branch ----
    gemm(x, Dd, 0, Wp, Dh, 0, 0, bp, tp, Dh, 0, NS, Dh, Dd, 0, 0, nullptr, 0, 1);
    rowsumsq_kernel<<<NS, 128, 0, stream>>>(tp, sq, Dh);
    gemm(tp, Dh, (long)S * Dh, tp, Dh, (long)S * Dh, 1, nullptr,
         d2, S, (long)S * S, S, S, Dh, 2, 0, sq, S, Bb);
    topk_kernel<<<NS / 8, 256, 0, stream>>>(d2, idxb, S, Kk);
    pool_kernel<<<NS, 256, 0, stream>>>(x, idxb, pooled, S, Dd, Kk);
    gemm(pooled, Dd, 0, We, Dd, 0, 0, be, topo, Dd, 0, NS, Dd, Dh, 0, 0, nullptr, 0, 1);
    ln_kernel<<<NS, 256, 0, stream>>>(topo, nullptr, ge, bbe, topo, Dd, 1);

    // ---- attention ----
    gemm(x, Dd, 0, Wq, Dd, 0, 0, bq, q,    Dd, 0, NS, Dd, Dd, 0, 0, nullptr, 0, 1);
    gemm(x, Dd, 0, Wk, Dd, 0, 0, bk, kbuf, Dd, 0, NS, Dd, Dd, 0, 0, nullptr, 0, 1);
    gemm(x, Dd, 0, Wv, Dd, 0, 0, bv, vbuf, Dd, 0, NS, Dd, Dd, 0, 0, nullptr, 0, 1);
    attn_kernel<<<dim3(S / 128, H, Bb), blk, 0, stream>>>(q, kbuf, vbuf, ao, S, Dd, HD);

    // ---- gate: gp = ao @ Wg[:D] + bg ; gp += topo @ Wg[D:] ----
    gemm(ao,   Dd, 0, Wg,                   Dd, 0, 0, bg,      gp, Dd, 0, NS, Dd, Dd, 0, 0, nullptr, 0, 1);
    gemm(topo, Dd, 0, Wg + (size_t)Dd * Dd, Dd, 0, 0, nullptr, gp, Dd, 0, NS, Dd, Dd, 0, 1, nullptr, 0, 1);
    gate_combine_kernel<<<(NS * Dd) / 256, blk, 0, stream>>>(gp, ao, topo, comb, (long)NS * Dd);
    gemm(comb, Dd, 0, Wo, Dd, 0, 0, bo, atn, Dd, 0, NS, Dd, Dd, 0, 0, nullptr, 0, 1);

    // ---- residual + LN + FFN ----
    ln_kernel<<<NS, 256, 0, stream>>>(x, atn, g1, b1, hbuf, Dd, 0);
    gemm(hbuf, Dd, 0, Wf1, Df, 0, 0, bf1, f1, Df, 0, NS, Df, Dd, 1, 0, nullptr, 0, 1);
    gemm(f1,   Df, 0, Wf2, Dd, 0, 0, bf2, f2, Dd, 0, NS, Dd, Df, 0, 0, nullptr, 0, 1);
    ln_kernel<<<NS, 256, 0, stream>>>(hbuf, f2, g2, b2, out, Dd, 0);
}
